// SingleDeterminantFermiNet_88227218195312
// MI455X (gfx1250) — matrix operand order repaired and checked by reading, not compiled, not run on hardware
//
#include <hip/hip_runtime.h>
#include <hip/hip_bf16.h>

// ---------------------------------------------------------------------------
// FermiNet forward on MI455X (gfx1250), one workgroup per batch element.
// Matmuls use v_wmma_f32_16x16x32_bf16 (bf16 in / f32 accumulate).
// All activations live in dynamic LDS (~182 KB of the 320 KB/WGP).
// Weights (~2.7MB f32 total) are streamed from global; they stay hot in the 192MB L2.
// Layer dims are template constants so all weight-load bounds checks fold.
// Transcendentals use the hardware v_exp/v_log/v_rcp paths (branchless).
// ---------------------------------------------------------------------------

typedef __attribute__((ext_vector_type(16))) __bf16 v16bf;
typedef __attribute__((ext_vector_type(8)))  float  v8f;

#define BATCH  1024
#define NELEC  32
#define NSPIN  16
#define NION   4
#define NDIM   3

// ----- LDS layout (float offsets) -----
#define OFF_H1   0                       // [32][256]   one-electron features
#define OFF_H2   (OFF_H1 + 32*256)       // [1024][32]  two-electron features
#define OFF_GC   (OFF_H2 + 1024*32)      // [32][64]    concat'd g (matches W_2e1e rows)
#define OFF_MIX  (OFF_GC + 32*64)        // [2][256]    mixed_up / mixed_dn
#define OFF_MU   (OFF_MIX + 2*256)       // [2][256]    mu / md
#define OFF_EL   (OFF_MU + 2*256)        // [32][3]     electron positions (padded)
#define OFF_ION  (OFF_EL + 128)          // [4][3]      ion positions (padded)
#define OFF_REI  (OFF_ION + 32)          // [32][4][3]  electron-ion displacements
#define OFF_ENV  (OFF_REI + 32*4*3)      // [2][16][16] envelope
#define OFF_DET  (OFF_ENV + 2*256)       // [2][16][16] determinant scratch
#define OFF_LOG  (OFF_DET + 2*256)       // [8]         per-spin log|det|
#define LDS_FLOATS (OFF_LOG + 8)
#define LDS_BYTES  (LDS_FLOATS * sizeof(float))

struct FNParams {
    const float* elec;
    const float* ion;
    const float* W2e[4];
    const float* W2e1e[4];
    const float* Wm[4];
    const float* Wu[4];
    const float* b1[4];
    const float* b2[4];
    const float* Wlin[2];
    const float* blin[2];
    const float* envA[2];
    const float* envion[2];
    float* out;
};

__device__ __forceinline__ __bf16 f2bf(float f) {
    return (__bf16)f;   // native v_cvt_pk_bf16_f32 on gfx1250
}

// Branchless tanh on the v_exp_f32 / v_rcp_f32 path:
// tanh(x) = sign(x)*(1-t)/(1+t), t = exp(-2|x|).
__device__ __forceinline__ float fast_tanh(float x) {
    const float ax = __builtin_fabsf(x);
    const float t  = __expf(-2.0f * ax);                       // v_exp_f32
    const float y  = (1.0f - t) * __builtin_amdgcn_rcpf(1.0f + t);  // v_rcp_f32
    return __builtin_copysignf(y, x);
}

__device__ __forceinline__ v8f wmma_bf16(v16bf a, v16bf b, v8f c) {
    return __builtin_amdgcn_wmma_f32_16x16x32_bf16(
        /*neg_a=*/false, a, /*neg_b=*/false, b,
        /*c_mod=*/(short)0, c, /*reuse_a=*/false, /*reuse_b=*/false);
}

// A fragment (16x32 bf16) from an LDS f32 matrix, per ISA 16-bit A layout:
// lanes 0-15: M=lane, K = {0..7,16..23}; lanes 16-31: M=lane-16, K = {8..15,24..31}
__device__ __forceinline__ v16bf load_A_lds(const float* src, int stride,
                                            int row0, int col0) {
    const int lane = threadIdx.x & 31;
    const int m    = row0 + (lane & 15);
    const int hi   = (lane >> 4) & 1;
    v16bf a;
#pragma unroll
    for (int p = 0; p < 8; ++p) {
        const int k = col0 + ((p < 4) ? (2 * p) : (16 + 2 * (p - 4))) + hi * 8;
        a[2 * p]     = f2bf(src[m * stride + k]);
        a[2 * p + 1] = f2bf(src[m * stride + k + 1]);
    }
    return a;
}

// B fragment (32x16 bf16) from global f32 weights (row-major KMAXxN, K-padded).
// lanes 0-15: N=lane, K = {0..7,16..23}; lanes 16-31: N=lane-16, K = {8..15,24..31}.
// KMAX is compile-time: full tiles fold to plain loads, short tiles to cndmask
// with clamped (always in-bounds) addresses — no exec-mask gymnastics.
template <int KMAX>
__device__ __forceinline__ v16bf load_B_gl(const float* __restrict__ W, int N,
                                           int k0, int n0) {
    const int lane = threadIdx.x & 31;
    const int n    = n0 + (lane & 15);
    const int hi   = (lane >> 4) & 1;
    v16bf b;
#pragma unroll
    for (int v = 0; v < 8; ++v) {
        const int k = k0 + ((v < 4) ? (2 * v) : (16 + 2 * (v - 4))) + hi * 8;
        if constexpr ((KMAX & 31) == 0) {
            // full K tile: unconditional loads (k0 constant after unroll)
            b[2 * v]     = f2bf(W[(size_t)k * N + n]);
            b[2 * v + 1] = f2bf(W[(size_t)(k + 1) * N + n]);
        } else {
            const bool in0 = (k < KMAX), in1 = (k + 1 < KMAX);
            const int ka = in0 ? k : 0;
            const int kb = in1 ? (k + 1) : 0;
            const float f0 = W[(size_t)ka * N + n];
            const float f1 = W[(size_t)kb * N + n];
            b[2 * v]     = f2bf(in0 ? f0 : 0.f);
            b[2 * v + 1] = f2bf(in1 ? f1 : 0.f);
        }
    }
    return b;
}

// ---------------------------------------------------------------------------
// One interaction layer.  D1/D2 = input widths (compile-time), RES = residual.
// ---------------------------------------------------------------------------
template <int D1, int D2, bool RES>
__device__ __forceinline__ void layer_body(
    float* h1f, float* h2f, float* gcat, float* mix, float* mumd,
    const float* __restrict__ Wu, const float* __restrict__ Wm,
    const float* __restrict__ W21, const float* __restrict__ W2,
    const float* __restrict__ b1v, const float* __restrict__ b2v,
    int tid, int wid, int lane) {

    // warm L2 for the big weight (global_prefetch path)
    __builtin_prefetch(Wu + (tid << 6), 0, 0);

    // (a) spin means mu / md over h1
    for (int c = tid; c < D1; c += 256) {
        float su = 0.f, sd = 0.f;
#pragma unroll
        for (int e = 0; e < 16; ++e) {
            su += h1f[e * 256 + c];
            sd += h1f[(16 + e) * 256 + c];
        }
        mumd[c]       = su * 0.0625f;
        mumd[256 + c] = sd * 0.0625f;
    }
    // (b) gcat: rows e<16 -> [g_up[e], g_dn[e]], rows e>=16 -> [g_dn[e], g_up[e]]
    if constexpr (2 * D2 < 64) {
        for (int t = tid; t < 32 * 64; t += 256) gcat[t] = 0.f;
    }
    __syncthreads();
    for (int t = tid; t < 32 * 2 * D2; t += 256) {
        const int e = t / (2 * D2);
        const int c = t % (2 * D2);
        const bool second = c >= D2;
        const int cf = second ? c - D2 : c;
        const int jbase = ((e < 16) ^ second) ? 0 : 16;  // g_up: j<16, g_dn: j>=16
        float s = 0.f;
#pragma unroll
        for (int j = 0; j < 16; ++j) s += h2f[(e * 32 + jbase + j) * 32 + cf];
        gcat[e * 64 + c] = s * 0.0625f;
    }
    // (c) mixed_up / mixed_dn = concat(mu,md) @ W_mixed (and swapped)
    {
        const int n = tid;  // 256 threads, 256 outputs
        float su = 0.f, sd = 0.f;
#pragma unroll 4
        for (int k = 0; k < D1; ++k) {
            const float w0 = Wm[(size_t)k * 256 + n];
            const float w1 = Wm[(size_t)(D1 + k) * 256 + n];
            const float vu = mumd[k], vd = mumd[256 + k];
            su += vu * w0 + vd * w1;
            sd += vd * w0 + vu * w1;
        }
        mix[n]       = su;
        mix[256 + n] = sd;
    }
    __syncthreads();

    // (d) h1' = tanh(h1@Wu + gcat@W21 + mixed + b1) [+ h1]
    constexpr int NK1 = (D1 + 31) / 32;
    constexpr int NK2 = (2 * D2 + 31) / 32;
    float newv[4][8];
#pragma unroll 1
    for (int tt = 0; tt < 4; ++tt) {
        const int tile = wid * 4 + tt;        // 32 tiles over 8 waves
        const int mt = tile & 1, nt = tile >> 1;
        v8f acc = {0.f, 0.f, 0.f, 0.f, 0.f, 0.f, 0.f, 0.f};
#pragma unroll
        for (int kk = 0; kk < NK1; ++kk) {
            v16bf a  = load_A_lds(h1f, 256, mt * 16, kk * 32);
            v16bf bb = load_B_gl<D1>(Wu, 256, kk * 32, nt * 16);
            acc = wmma_bf16(a, bb, acc);
        }
#pragma unroll
        for (int kk = 0; kk < NK2; ++kk) {
            v16bf a  = load_A_lds(gcat, 64, mt * 16, kk * 32);
            v16bf bb = load_B_gl<2 * D2>(W21, 256, kk * 32, nt * 16);
            acc = wmma_bf16(a, bb, acc);
        }
        const int n     = nt * 16 + (lane & 15);
        const int mbase = mt * 16 + ((lane >> 4) & 1) * 8;
#pragma unroll
        for (int j = 0; j < 8; ++j) {
            const int m = mbase + j;
            float v = acc[j] + mix[(m < 16 ? 0 : 256) + n] + b1v[n];
            v = fast_tanh(v);
            if constexpr (RES) v += h1f[m * 256 + n];   // residual (reads OLD h1)
            newv[tt][j] = v;
        }
    }
    __syncthreads();   // all reads of old h1 done
#pragma unroll
    for (int tt = 0; tt < 4; ++tt) {
        const int tile = wid * 4 + tt;
        const int mt = tile & 1, nt = tile >> 1;
        const int n     = nt * 16 + (lane & 15);
        const int mbase = mt * 16 + ((lane >> 4) & 1) * 8;
#pragma unroll
        for (int j = 0; j < 8; ++j) h1f[(mbase + j) * 256 + n] = newv[tt][j];
    }
    __syncthreads();

    // (e) h2' = tanh(h2@W2 + b2) [+ h2]  — in place, wave-owned M row groups
#pragma unroll 1
    for (int mtt = 0; mtt < 8; ++mtt) {
        const int mt = wid * 8 + mtt;   // 64 M tiles over 8 waves
        v16bf a = load_A_lds(h2f, 32, mt * 16, 0);
        v8f acc0 = {0.f, 0.f, 0.f, 0.f, 0.f, 0.f, 0.f, 0.f};
        v8f acc1 = {0.f, 0.f, 0.f, 0.f, 0.f, 0.f, 0.f, 0.f};
        v16bf bf0 = load_B_gl<D2>(W2, 32, 0, 0);
        v16bf bf1 = load_B_gl<D2>(W2, 32, 0, 16);
        acc0 = wmma_bf16(a, bf0, acc0);
        acc1 = wmma_bf16(a, bf1, acc1);
        const int nn    = lane & 15;
        const int mbase = mt * 16 + ((lane >> 4) & 1) * 8;
        float o0[8], o1[8];
#pragma unroll
        for (int j = 0; j < 8; ++j) {
            const int m = mbase + j;
            float v0 = fast_tanh(acc0[j] + b2v[nn]);
            float v1 = fast_tanh(acc1[j] + b2v[16 + nn]);
            if constexpr (RES) { v0 += h2f[m * 32 + nn]; v1 += h2f[m * 32 + 16 + nn]; }
            o0[j] = v0; o1[j] = v1;
        }
#pragma unroll
        for (int j = 0; j < 8; ++j) {
            const int m = mbase + j;
            h2f[m * 32 + nn]      = o0[j];
            h2f[m * 32 + 16 + nn] = o1[j];
        }
    }
    __syncthreads();
}

extern "C" __global__ void __launch_bounds__(256, 1)
ferminet_fwd_kernel(FNParams P) {
    extern __shared__ float smem[];
    float* h1f  = smem + OFF_H1;   // [32][256]
    float* h2f  = smem + OFF_H2;   // [1024][32]
    float* gcat = smem + OFF_GC;   // [32][64]
    float* mix  = smem + OFF_MIX;  // [2][256]
    float* mumd = smem + OFF_MU;   // [2][256]
    float* el   = smem + OFF_EL;   // [32][3]
    float* ionp = smem + OFF_ION;  // [4][3]
    float* rei  = smem + OFF_REI;  // [32][4][3]
    float* envb = smem + OFF_ENV;  // [2][16][16]
    float* dets = smem + OFF_DET;  // [2][16][16]
    float* logp = smem + OFF_LOG;

    const int tid  = threadIdx.x;
    const int wid  = tid >> 5;
    const int lane = tid & 31;
    const int b    = blockIdx.x;

    // ---------------- Stage 0: inputs, h1, h2 ----------------
    for (int t = tid; t < NELEC * NDIM; t += 256) el[t] = P.elec[(size_t)b * NELEC * NDIM + t];
    for (int t = tid; t < NION * NDIM; t += 256)  ionp[t] = P.ion[t];
    __syncthreads();

    for (int t = tid; t < 32 * 256; t += 256) h1f[t] = 0.f;   // zero-pad for K=16 layer
    __syncthreads();

    // h1 = [r_ei, |r_ei|] per (electron, ion)
    for (int t = tid; t < NELEC * NION; t += 256) {
        const int e = t >> 2, a = t & 3;
        const float dx = el[e * 3 + 0] - ionp[a * 3 + 0];
        const float dy = el[e * 3 + 1] - ionp[a * 3 + 1];
        const float dz = el[e * 3 + 2] - ionp[a * 3 + 2];
        const float nr = sqrtf(dx * dx + dy * dy + dz * dz);
        rei[(e * 4 + a) * 3 + 0] = dx;
        rei[(e * 4 + a) * 3 + 1] = dy;
        rei[(e * 4 + a) * 3 + 2] = dz;
        h1f[e * 256 + a * 4 + 0] = dx;
        h1f[e * 256 + a * 4 + 1] = dy;
        h1f[e * 256 + a * 4 + 2] = dz;
        h1f[e * 256 + a * 4 + 3] = nr;
    }
    // h2 = [r_ee, |r_ee|] per pair; cols 4..31 zeroed (K padding for layer 0)
    for (int t = tid; t < NELEC * NELEC; t += 256) {
        const int i = t >> 5, j = t & 31;
        const float dx = el[i * 3 + 0] - el[j * 3 + 0];
        const float dy = el[i * 3 + 1] - el[j * 3 + 1];
        const float dz = el[i * 3 + 2] - el[j * 3 + 2];
        const float n2 = dx * dx + dy * dy + dz * dz;
        float* row = h2f + t * 32;
        row[0] = dx; row[1] = dy; row[2] = dz;
        row[3] = (i == j) ? 0.f : sqrtf(n2);
#pragma unroll
        for (int c = 4; c < 32; ++c) row[c] = 0.f;
    }
    __syncthreads();

    // ---------------- 4 interaction layers ----------------
    layer_body<16, 4, false>(h1f, h2f, gcat, mix, mumd,
                             P.Wu[0], P.Wm[0], P.W2e1e[0], P.W2e[0],
                             P.b1[0], P.b2[0], tid, wid, lane);
#pragma unroll 1
    for (int l = 1; l < 4; ++l) {
        layer_body<256, 32, true>(h1f, h2f, gcat, mix, mumd,
                                  P.Wu[l], P.Wm[l], P.W2e1e[l], P.W2e[l],
                                  P.b1[l], P.b2[l], tid, wid, lane);
    }

    // ---------------- Orbitals + envelope + slogdet ----------------
    // env[s][i][o] = sum_a env_ion[a][o] * exp(-|env_A[a][o] @ r_ei[s*16+i][a]|)
    for (int t = tid; t < 512; t += 256) {
        const int s = t >> 8, rem = t & 255;
        const int i = rem >> 4, o = rem & 15;
        const float* __restrict__ eA = P.envA[s];
        const float* __restrict__ eI = P.envion[s];
        const int eg = s * 16 + i;
        float envv = 0.f;
#pragma unroll
        for (int a = 0; a < NION; ++a) {
            const float* r3 = rei + (eg * 4 + a) * 3;
            float d2s = 0.f;
#pragma unroll
            for (int e = 0; e < 3; ++e) {
                const float* Ar = eA + (((a * 16 + o) * 3 + e) * 3);
                const float sc = Ar[0] * r3[0] + Ar[1] * r3[1] + Ar[2] * r3[2];
                d2s += sc * sc;
            }
            envv += eI[a * 16 + o] * __expf(-sqrtf(d2s));   // v_exp_f32 path
        }
        envb[t] = envv;
    }
    // lin = h1[spin] @ W_lin  (one wave per spin; 8 WMMA each)
    v8f linacc = {0.f, 0.f, 0.f, 0.f, 0.f, 0.f, 0.f, 0.f};
    if (wid < 2) {
        const int s = wid;
#pragma unroll
        for (int kk = 0; kk < 8; ++kk) {
            v16bf a  = load_A_lds(h1f, 256, s * 16, kk * 32);
            v16bf bb = load_B_gl<256>(P.Wlin[s], 16, kk * 32, 0);
            linacc = wmma_bf16(a, bb, linacc);
        }
    }
    __syncthreads();   // envelope ready
    if (wid < 2) {
        const int s = wid;
        const int n     = lane & 15;
        const int mbase = ((lane >> 4) & 1) * 8;
        const float* __restrict__ bl = P.blin[s];
#pragma unroll
        for (int j = 0; j < 8; ++j) {
            const int m = mbase + j;
            dets[s * 256 + m * 16 + n] = (linacc[j] + bl[n]) * envb[s * 256 + m * 16 + n];
        }
    }
    __syncthreads();

    // log|det| via partially pivoted LU — one wave per spin, matrix in LDS
    if (wid < 2) {
        volatile float* M = dets + wid * 256;
        float lacc = 0.f;
        for (int k = 0; k < 16; ++k) {
            int piv = k;
            float best = __builtin_fabsf(M[k * 16 + k]);
            for (int r = k + 1; r < 16; ++r) {
                const float v = __builtin_fabsf(M[r * 16 + k]);
                if (v > best) { best = v; piv = r; }
            }
            if (piv != k && lane < 16) {
                const float tmp = M[k * 16 + lane];
                M[k * 16 + lane]   = M[piv * 16 + lane];
                M[piv * 16 + lane] = tmp;
            }
            __builtin_amdgcn_wave_barrier();
            const float pv = M[k * 16 + k];
            if (lane == 0) lacc += __logf(__builtin_fabsf(pv) + 1e-30f);  // v_log_f32
            if (lane > k && lane < 16) {
                const float f = M[lane * 16 + k] / pv;
                for (int j = k; j < 16; ++j) M[lane * 16 + j] -= f * M[k * 16 + j];
            }
            __builtin_amdgcn_wave_barrier();
        }
        if (lane == 0) logp[wid] = lacc;
    }
    __syncthreads();
    if (tid == 0) P.out[b] = logp[0] + logp[1];
}

extern "C" void kernel_launch(void* const* d_in, const int* in_sizes, int n_in,
                              void* d_out, int out_size, void* d_ws, size_t ws_size,
                              hipStream_t stream) {
    (void)in_sizes; (void)n_in; (void)out_size; (void)d_ws; (void)ws_size;

    FNParams p;
    p.elec = (const float*)d_in[0];
    p.ion  = (const float*)d_in[1];
    // params flattened sorted: layers (W_2e, W_2e1e, W_mixed, W_unmixed, b_1e, b_2e) x4,
    // then orbitals (W_lin, b_lin, env_A, env_ion) x2
    for (int l = 0; l < 4; ++l) {
        const int base = 2 + l * 6;
        p.W2e[l]   = (const float*)d_in[base + 0];
        p.W2e1e[l] = (const float*)d_in[base + 1];
        p.Wm[l]    = (const float*)d_in[base + 2];
        p.Wu[l]    = (const float*)d_in[base + 3];
        p.b1[l]    = (const float*)d_in[base + 4];
        p.b2[l]    = (const float*)d_in[base + 5];
    }
    for (int s = 0; s < 2; ++s) {
        const int base = 26 + s * 4;
        p.Wlin[s]   = (const float*)d_in[base + 0];
        p.blin[s]   = (const float*)d_in[base + 1];
        p.envA[s]   = (const float*)d_in[base + 2];
        p.envion[s] = (const float*)d_in[base + 3];
    }
    p.out = (float*)d_out;

    hipLaunchKernelGGL(ferminet_fwd_kernel, dim3(BATCH), dim3(256), LDS_BYTES,
                       stream, p);
}
